// RBFKernel_65481071400245
// MI455X (gfx1250) — compile-verified, matrix operand order checked by reading, not run on hardware
//
#include <hip/hip_runtime.h>
#include <hip/hip_bf16.h>
#include <math.h>

// ---------------------------------------------------------------------------
// RBF kernel: out[n,m] = exp(-||xhat_n - yhat_m||^2), xhat/yhat column-normalized.
// Pipeline: col-stats -> normalize + bf16 hi/lo split pack + rownorm ->
//           bf16x3 WMMA GEMM (16x16x32_bf16) with fused exp epilogue.
// Target: gfx1250 (CDNA5), wave32.
//
// bf16x3: xy = xh*yh + xh*yl + xl*yh (lo*lo dropped, ~1e-4 abs error on D=512
// dot products) -> near-f32 accuracy at 8x K-depth per WMMA vs f32 16x16x4.
// ---------------------------------------------------------------------------

typedef __attribute__((ext_vector_type(16))) __bf16 v16bf;
typedef __attribute__((ext_vector_type(8)))  __bf16 v8bf;
typedef __attribute__((ext_vector_type(8)))  float  v8f;

#define N_ROWS 8192
#define N_DIM  512
#define PK_STRIDE 1024          // per-row packed layout: hi[512] | lo[512]
#define GAMMA_F 1.0f
#define EPS_F   1e-8f

// ---------------------------------------------------------------------------
// Stage 1a: partial column sums/sumsq. grid=(4,64): blockIdx.x selects
// {matrix, col-half}; blockIdx.y a 128-row chunk. Fully coalesced.
// ---------------------------------------------------------------------------
__global__ __launch_bounds__(256)
void colstats_partial(const float* __restrict__ x, const float* __restrict__ y,
                      float* __restrict__ psum, float* __restrict__ psq) {
    const int mat = blockIdx.x >> 1;
    const int col = ((blockIdx.x & 1) << 8) + threadIdx.x;
    const float* __restrict__ src = mat ? y : x;
    const int r0 = blockIdx.y * 128;
    float s = 0.f, q = 0.f;
    #pragma unroll 4
    for (int r = 0; r < 128; ++r) {
        float v = src[(size_t)(r0 + r) * N_DIM + col];
        s += v;
        q += v * v;
    }
    const int idx = (mat * 64 + blockIdx.y) * N_DIM + col;
    psum[idx] = s;
    psq[idx]  = q;
}

// ---------------------------------------------------------------------------
// Stage 1b: finish column stats. mean, unbiased std (ddof=1), inv = 1/(std+eps).
// ---------------------------------------------------------------------------
__global__ __launch_bounds__(256)
void colstats_final(const float* __restrict__ psum, const float* __restrict__ psq,
                    float* __restrict__ meanArr, float* __restrict__ invArr) {
    const int mat = blockIdx.x >> 1;
    const int col = ((blockIdx.x & 1) << 8) + threadIdx.x;
    float s = 0.f, q = 0.f;
    for (int c = 0; c < 64; ++c) {
        const int idx = (mat * 64 + c) * N_DIM + col;
        s += psum[idx];
        q += psq[idx];
    }
    const float mean = s * (1.0f / (float)N_ROWS);
    const float var  = (q - (float)N_ROWS * mean * mean) * (1.0f / (float)(N_ROWS - 1));
    const float sd   = sqrtf(fmaxf(var, 0.0f));
    meanArr[mat * N_DIM + col] = mean;
    invArr[mat * N_DIM + col]  = 1.0f / (sd + EPS_F);
}

// ---------------------------------------------------------------------------
// Stage 2: normalize, split into bf16 hi/lo, pack rows as [hi[512] | lo[512]],
// and wave-reduce exact f32 row norms. One wave32 per row.
// ---------------------------------------------------------------------------
__global__ __launch_bounds__(128)
void normalize_pack(const float* __restrict__ x, const float* __restrict__ y,
                    const float* __restrict__ meanArr, const float* __restrict__ invArr,
                    __bf16* __restrict__ xpk, __bf16* __restrict__ ypk,
                    float* __restrict__ x2, float* __restrict__ y2) {
    const int lane = threadIdx.x & 31;
    const int rowg = blockIdx.x * 4 + (threadIdx.x >> 5);
    const int mat  = rowg >> 13;           // 0 -> x, 1 -> y
    const int row  = rowg & (N_ROWS - 1);
    const float* __restrict__ src = mat ? y : x;
    __bf16* __restrict__ dst = mat ? ypk : xpk;
    float* __restrict__ nrm = mat ? y2 : x2;
    const float* __restrict__ mp = meanArr + mat * N_DIM;
    const float* __restrict__ ip = invArr + mat * N_DIM;

    float acc = 0.f;
    #pragma unroll
    for (int i = 0; i < N_DIM / 32; ++i) {
        const int c = lane + i * 32;
        const float v = (src[(size_t)row * N_DIM + c] - mp[c]) * ip[c];
        const __bf16 h = (__bf16)v;              // RNE
        const __bf16 l = (__bf16)(v - (float)h); // residual
        dst[(size_t)row * PK_STRIDE + c]         = h;
        dst[(size_t)row * PK_STRIDE + N_DIM + c] = l;
        acc += v * v;                            // exact f32 row norm
    }
    #pragma unroll
    for (int off = 16; off > 0; off >>= 1)
        acc += __shfl_xor(acc, off, 32);
    if (lane == 0) nrm[row] = acc;
}

// ---------------------------------------------------------------------------
// bf16 16x32 A fragment load (ISA 7.12.2, 16-bit A layout):
//   lane16 = M row (A) / N col (B, mirrored); hk = lane>>4 picks K-chunk.
//   v16bf elements 0..7  = K [k + hk*8 .. +7]        (VGPR 0-3)
//   v16bf elements 8..15 = K [k + 16 + hk*8 .. +7]   (VGPR 4-7)
//   -> two contiguous 16-byte runs from a row-major bf16 row.
// `base` already includes row*PK_STRIDE + hk*8.
// ---------------------------------------------------------------------------
__device__ __forceinline__ v16bf load_frag(const __bf16* __restrict__ base, int off) {
    union { v16bf v; struct { v8bf a, b; } h; } u;
    u.h.a = *(const v8bf*)(base + off);
    u.h.b = *(const v8bf*)(base + off + 16);
    return u.v;
}

__device__ __forceinline__ v8f wmma_bf16(v16bf a, v16bf b, v8f c) {
    return __builtin_amdgcn_wmma_f32_16x16x32_bf16(
        /*neg_a=*/false, a, /*neg_b=*/false, b,
        /*c_mod=*/(short)0, c, /*reuse_a=*/false, /*reuse_b=*/false);
}

// ---------------------------------------------------------------------------
// Stage 3: bf16x3 WMMA GEMM + RBF epilogue.
// Block = 256 threads = 8 waves (2x4). Wave tile = 64x64 (4x4 WMMA tiles).
// Per K=32 step: A hi/lo fragments resident (4 tiles x 2), B streamed per ni.
// 48 v_wmma_f32_16x16x32_bf16 per K-step per wave.
// C/D layout: VGPR r, lanes 0-15 -> M=r, lanes 16-31 -> M=r+8, N=lane16.
// ---------------------------------------------------------------------------
__global__ __launch_bounds__(256)
void rbf_gemm_bf16x3(const __bf16* __restrict__ xpk, const __bf16* __restrict__ ypk,
                     const float* __restrict__ x2, const float* __restrict__ y2,
                     float* __restrict__ out) {
    const int lane   = threadIdx.x & 31;
    const int wave   = threadIdx.x >> 5;     // 0..7
    const int lane16 = lane & 15;
    const int hk     = lane >> 4;            // K-chunk select

    const int rowBase = blockIdx.x * 128 + (wave >> 2) * 64;   // M origin
    const int colBase = blockIdx.y * 256 + (wave & 3) * 64;    // N origin

    v8f acc[4][4];
    #pragma unroll
    for (int mi = 0; mi < 4; ++mi)
        #pragma unroll
        for (int ni = 0; ni < 4; ++ni)
            acc[mi][ni] = (v8f){0.f, 0.f, 0.f, 0.f, 0.f, 0.f, 0.f, 0.f};

    const __bf16* aBase[4];
    const __bf16* bBase[4];
    #pragma unroll
    for (int mi = 0; mi < 4; ++mi)
        aBase[mi] = xpk + (size_t)(rowBase + mi * 16 + lane16) * PK_STRIDE + hk * 8;
    #pragma unroll
    for (int ni = 0; ni < 4; ++ni)
        bBase[ni] = ypk + (size_t)(colBase + ni * 16 + lane16) * PK_STRIDE + hk * 8;

    #pragma unroll 1
    for (int k = 0; k < N_DIM; k += 32) {
        v16bf ah[4], al[4];
        #pragma unroll
        for (int mi = 0; mi < 4; ++mi) {
            ah[mi] = load_frag(aBase[mi], k);
            al[mi] = load_frag(aBase[mi], N_DIM + k);   // lo region at +512
        }
        #pragma unroll
        for (int ni = 0; ni < 4; ++ni) {
            const v16bf bh = load_frag(bBase[ni], k);
            const v16bf bl = load_frag(bBase[ni], N_DIM + k);
            // three passes so consecutive WMMAs hit different accumulators
            #pragma unroll
            for (int mi = 0; mi < 4; ++mi)
                acc[mi][ni] = wmma_bf16(ah[mi], bh, acc[mi][ni]);
            #pragma unroll
            for (int mi = 0; mi < 4; ++mi)
                acc[mi][ni] = wmma_bf16(ah[mi], bl, acc[mi][ni]);
            #pragma unroll
            for (int mi = 0; mi < 4; ++mi)
                acc[mi][ni] = wmma_bf16(al[mi], bh, acc[mi][ni]);
        }
    }

    // Epilogue: out = exp(-gamma * max(x2 + y2 - 2*xy, 0)), NT stores
    // (256MB output must not evict the 32MB packed operands from L2).
    #pragma unroll
    for (int mi = 0; mi < 4; ++mi) {
        float x2v[8];
        #pragma unroll
        for (int r = 0; r < 8; ++r)
            x2v[r] = x2[rowBase + mi * 16 + hk * 8 + r];
        #pragma unroll
        for (int ni = 0; ni < 4; ++ni) {
            const int col = colBase + ni * 16 + lane16;
            const float y2v = y2[col];
            #pragma unroll
            for (int r = 0; r < 8; ++r) {
                const int row = rowBase + mi * 16 + hk * 8 + r;
                const float d = fmaxf(x2v[r] + y2v - 2.0f * acc[mi][ni][r], 0.0f);
                __builtin_nontemporal_store(__expf(-GAMMA_F * d),
                                            &out[(size_t)row * N_ROWS + col]);
            }
        }
    }
}

// ---------------------------------------------------------------------------
// Host launcher. Workspace layout:
//   xpk [8192*1024 bf16 = 16MB] | ypk [16MB] | x2 [8192 f32] | y2 [8192 f32] |
//   mean [2*512] | inv [2*512] | psum [2*64*512] | psq [2*64*512]
// Total ~32.6 MB.
// ---------------------------------------------------------------------------
extern "C" void kernel_launch(void* const* d_in, const int* in_sizes, int n_in,
                              void* d_out, int out_size, void* d_ws, size_t ws_size,
                              hipStream_t stream) {
    const float* x = (const float*)d_in[0];
    const float* y = (const float*)d_in[1];
    float* out = (float*)d_out;

    __bf16* xpk = (__bf16*)d_ws;
    __bf16* ypk = xpk + (size_t)N_ROWS * PK_STRIDE;
    float* x2      = (float*)(ypk + (size_t)N_ROWS * PK_STRIDE);
    float* y2      = x2 + N_ROWS;
    float* meanArr = y2 + N_ROWS;
    float* invArr  = meanArr + 2 * N_DIM;
    float* psum    = invArr + 2 * N_DIM;
    float* psq     = psum + 2 * 64 * N_DIM;

    colstats_partial<<<dim3(4, 64), 256, 0, stream>>>(x, y, psum, psq);
    colstats_final<<<dim3(4), 256, 0, stream>>>(psum, psq, meanArr, invArr);
    normalize_pack<<<dim3((2 * N_ROWS) / 4), 128, 0, stream>>>(
        x, y, meanArr, invArr, xpk, ypk, x2, y2);
    rbf_gemm_bf16x3<<<dim3(N_ROWS / 128, N_ROWS / 256), 256, 0, stream>>>(
        xpk, ypk, x2, y2, out);
}
